// DuelingDQN_12489764897161
// MI455X (gfx1250) — matrix operand order repaired and checked by reading, not compiled, not run on hardware
//
#include <hip/hip_runtime.h>
#include <hip/hip_bf16.h>

// ---------------------------------------------------------------------------
// Dueling DQN fused MLP for gfx1250 (MI455X).
//  - All GEMMs via V_WMMA_F32_16X16X32_F16 (f16 inputs, f32 accumulate).
//  - Weights pre-packed into WMMA B-fragment layout (L2-resident, ~396 KB).
//  - Per-wave 16-row tiles, activations in private LDS panels, zero barriers.
//  - LayerNorm computed on f32 accumulators with wave32 shfl_xor reductions.
// ---------------------------------------------------------------------------

typedef __attribute__((ext_vector_type(16))) _Float16 v16h;
typedef __attribute__((ext_vector_type(8)))  _Float16 v8h;
typedef __attribute__((ext_vector_type(8)))  float    v8f;

#define NB       131072     // batch
#define SLEN     199        // state length (D_IN = 200 incl. time_delta)
#define NE       3          // event heads
#define NA       32         // actions

union V16U { v16h v; v8h h[2]; };
static __device__ __forceinline__ v16h mk16(v8h lo, v8h hi) {
    V16U u; u.h[0] = lo; u.h[1] = hi; return u.v;
}

// ---------------------------------------------------------------------------
// GEMM: 16 rows (one wave) x (NT*16) cols, K reduction, A from LDS (stride K),
// B fragments streamed from pre-packed global (L2).
// A-fragment k order used here (contiguous, same as B pack; not the ISA table order):
//   lanes 0-15 : row=lane,  K = {kt*32+0..15}
//   lanes 16-31: row=l-16,  K = {kt*32+16..31}
// B-fragment packed so each lane reads 16 contiguous halves (32 B).
// ---------------------------------------------------------------------------
template<int K, int NT>
static __device__ __forceinline__ void gemm_tile(const _Float16* aLds,
                                                 const _Float16* __restrict__ wPack,
                                                 v8f* acc, int lane)
{
    const int hf = lane >> 4;
    const int lr = lane & 15;
#pragma unroll
    for (int kt = 0; kt < K / 32; ++kt) {
        const _Float16* ap = aLds + lr * K + kt * 32 + hf * 16;
        v16h a = mk16(*(const v8h*)(ap), *(const v8h*)(ap + 8));
#pragma unroll
        for (int nt = 0; nt < NT; ++nt) {
            const _Float16* bp = wPack + (((kt * NT + nt) * 32 + lane) << 4);
            v16h b = mk16(*(const v8h*)(bp), *(const v8h*)(bp + 8));
            acc[nt] = __builtin_amdgcn_wmma_f32_16x16x32_f16(
                false, a, false, b, (short)0, acc[nt], false, false);
        }
    }
}

// ---------------------------------------------------------------------------
// bias + LayerNorm + ReLU on f32 accumulators, write f16 back to LDS.
// C/D layout: acc[nt][i] = row (i + 8*(lane>=16)), col nt*16 + (lane&15).
// Row stats reduced across the 16-lane half via shfl_xor (wave32).
// ---------------------------------------------------------------------------
template<int NT>
static __device__ __forceinline__ void bias_ln_relu_store(
    v8f* acc,
    const float* __restrict__ bias, const float* __restrict__ g,
    const float* __restrict__ be,
    _Float16* outLds, int stride, int lane)
{
    const int hf  = lane >> 4;
    const int col = lane & 15;
    float cb[NT], cg[NT], cbe[NT];
#pragma unroll
    for (int nt = 0; nt < NT; ++nt) {
        cb[nt]  = bias[nt * 16 + col];
        cg[nt]  = g[nt * 16 + col];
        cbe[nt] = be[nt * 16 + col];
    }
#pragma unroll
    for (int nt = 0; nt < NT; ++nt)
#pragma unroll
        for (int i = 0; i < 8; ++i) acc[nt][i] += cb[nt];

    const float invN = 1.0f / (float)(NT * 16);
#pragma unroll
    for (int i = 0; i < 8; ++i) {
        float s = 0.f, q = 0.f;
#pragma unroll
        for (int nt = 0; nt < NT; ++nt) { float v = acc[nt][i]; s += v; q += v * v; }
#pragma unroll
        for (int m = 1; m < 16; m <<= 1) {
            s += __shfl_xor(s, m, 32);
            q += __shfl_xor(q, m, 32);
        }
        const float mu = s * invN;
        const float rs = rsqrtf(q * invN - mu * mu + 1e-5f);
        const int row = i + 8 * hf;
#pragma unroll
        for (int nt = 0; nt < NT; ++nt) {
            float v = (acc[nt][i] - mu) * rs * cg[nt] + cbe[nt];
            v = v > 0.f ? v : 0.f;
            outLds[row * stride + nt * 16 + col] = (_Float16)v;
        }
    }
}

// ---------------------------------------------------------------------------
// Weight pre-pack: f32 (K x N row-major) -> f16 WMMA B-fragment tiles.
// packed[((kt*NT+nt)*32 + lane)*16 + j] = W[kt*32 + (lane>>4)*16 + j][nt*16 + lane&15]
// k >= realK zero-padded (covers W1's 200 -> 256 K pad).
// ---------------------------------------------------------------------------
__global__ void pack_b_frags(const float* __restrict__ W, _Float16* __restrict__ dst,
                             int KT, int NT, int N, int realK, int strideW, int nBatch)
{
    const int per = KT * NT * 512;
    const long total = (long)per * nBatch;
    const long idx = (long)blockIdx.x * blockDim.x + threadIdx.x;
    if (idx >= total) return;
    const int b    = (int)(idx / per);
    const int r    = (int)(idx - (long)b * per);
    const int j    = r & 15;
    const int l    = (r >> 4) & 31;
    const int tile = r >> 9;
    const int nt   = tile % NT;
    const int kt   = tile / NT;
    const int k    = kt * 32 + (l >> 4) * 16 + j;
    const int n    = nt * 16 + (l & 15);
    const float v  = (k < realK) ? W[(long)b * strideW + (long)k * N + n] : 0.f;
    dst[idx] = (_Float16)v;
}

// ---------------------------------------------------------------------------
// Fused forward. Block = 128 threads = 4 wave32; each wave owns 16 rows.
// ---------------------------------------------------------------------------
__global__ __launch_bounds__(128)
void dueling_fused(const float* __restrict__ state, const float* __restrict__ td,
                   const float* __restrict__ b1, const float* __restrict__ g1, const float* __restrict__ be1,
                   const float* __restrict__ b2, const float* __restrict__ g2, const float* __restrict__ be2,
                   const float* __restrict__ b3, const float* __restrict__ g3, const float* __restrict__ be3,
                   const float* __restrict__ bv1, const float* __restrict__ gv, const float* __restrict__ bev,
                   const float* __restrict__ Wv2, const float* __restrict__ bv2,
                   const float* __restrict__ ba1, const float* __restrict__ ga, const float* __restrict__ bea,
                   const float* __restrict__ ba2, const int* __restrict__ ev,
                   const _Float16* __restrict__ W1p, const _Float16* __restrict__ W2p,
                   const _Float16* __restrict__ W3p, const _Float16* __restrict__ Wv1p,
                   const _Float16* __restrict__ Wa1p, const _Float16* __restrict__ Wa2p,
                   float* __restrict__ out)
{
    __shared__ _Float16 sA[4 * 16 * 256];   // per-wave activation panel (32 KB)
    __shared__ _Float16 sB[4 * 16 * 64];    // per-wave head panel       (8 KB)

    const int lane = threadIdx.x & 31;
    const int wv   = threadIdx.x >> 5;
    _Float16* bufA = sA + wv * 16 * 256;
    _Float16* bufB = sB + wv * 16 * 64;
    const int rowbase = blockIdx.x * 64 + wv * 16;

    // ---- load x = [state | td | zero-pad] into bufA as f16, stride 256 ----
#pragma unroll 4
    for (int t = 0; t < 128; ++t) {
        const int linear = t * 32 + lane;
        const int r = linear >> 8;
        const int c = linear & 255;
        const int gr = rowbase + r;
        float v = 0.f;
        if (c < SLEN)       v = state[gr * SLEN + c];
        else if (c == SLEN) v = td[gr];
        bufA[r * 256 + c] = (_Float16)v;
    }

    // Keep downstream weights warm in L2 while layer 1 runs.
    __builtin_prefetch(W2p, 0, 1);
    __builtin_prefetch(W3p, 0, 1);

    // ---- layer 1: x(16x256pad) @ W1(256x256) ----
    {
        v8f acc[16] = {};
        gemm_tile<256, 16>(bufA, W1p, acc, lane);
        bias_ln_relu_store<16>(acc, b1, g1, be1, bufA, 256, lane);
    }
    // ---- layer 2: h(16x256) @ W2(256x256) ----
    {
        v8f acc[16] = {};
        gemm_tile<256, 16>(bufA, W2p, acc, lane);
        bias_ln_relu_store<16>(acc, b2, g2, be2, bufA, 256, lane);
    }
    // ---- layer 3: h(16x256) @ W3(256x128) -> f ----
    {
        v8f acc[8] = {};
        gemm_tile<256, 8>(bufA, W3p, acc, lane);
        bias_ln_relu_store<8>(acc, b3, g3, be3, bufA, 128, lane);
    }

    // ---- value head: f(16x128) @ Wv1(128x64) -> LN/ReLU -> @ Wv2(64x1) ----
    float vrow = 0.f;
    {
        v8f acc[4] = {};
        gemm_tile<128, 4>(bufA, Wv1p, acc, lane);
        bias_ln_relu_store<4>(acc, bv1, gv, bev, bufB, 64, lane);
        if (lane < 16) {
            float s = bv2[0];
#pragma unroll 8
            for (int k = 0; k < 64; ++k)
                s += (float)bufB[lane * 64 + k] * Wv2[k];
            vrow = s;
        }
    }

    const int hf  = lane >> 4;
    const int col = lane & 15;

    // ---- advantage heads: all 3, select per-row by event_type ----
#pragma unroll 1
    for (int e = 0; e < NE; ++e) {
        v8f acc1[4] = {};
        gemm_tile<128, 4>(bufA, Wa1p + e * (4 * 4 * 512), acc1, lane);
        bias_ln_relu_store<4>(acc1, ba1 + e * 64, ga + e * 64, bea + e * 64,
                              bufB, 64, lane);
        v8f acc2[2] = {};
        gemm_tile<64, 2>(bufB, Wa2p + e * (2 * 2 * 512), acc2, lane);

        const float c0 = ba2[e * NA + col];
        const float c1 = ba2[e * NA + 16 + col];
#pragma unroll
        for (int i = 0; i < 8; ++i) {
            float a0 = acc2[0][i] + c0;
            float a1 = acc2[1][i] + c1;
            float s  = a0 + a1;
#pragma unroll
            for (int m = 1; m < 16; m <<= 1) s += __shfl_xor(s, m, 32);
            const float mean = s * (1.0f / 32.0f);
            const int row = i + 8 * hf;
            const int gr  = rowbase + row;
            const float vm = __shfl(vrow, row, 32);
            if (ev[gr] == e) {
                out[gr * NA + col]      = vm + a0 - mean;
                out[gr * NA + 16 + col] = vm + a1 - mean;
            }
        }
    }
}

// ---------------------------------------------------------------------------
// host launch
// ---------------------------------------------------------------------------
extern "C" void kernel_launch(void* const* d_in, const int* in_sizes, int n_in,
                              void* d_out, int out_size, void* d_ws, size_t ws_size,
                              hipStream_t stream)
{
    (void)in_sizes; (void)n_in; (void)out_size; (void)ws_size;

    const float* state = (const float*)d_in[0];
    const float* td    = (const float*)d_in[1];
    const float* W1    = (const float*)d_in[2];
    const float* b1    = (const float*)d_in[3];
    const float* g1    = (const float*)d_in[4];
    const float* be1   = (const float*)d_in[5];
    const float* W2    = (const float*)d_in[6];
    const float* b2    = (const float*)d_in[7];
    const float* g2    = (const float*)d_in[8];
    const float* be2   = (const float*)d_in[9];
    const float* W3    = (const float*)d_in[10];
    const float* b3    = (const float*)d_in[11];
    const float* g3    = (const float*)d_in[12];
    const float* be3   = (const float*)d_in[13];
    const float* Wv1   = (const float*)d_in[14];
    const float* bv1   = (const float*)d_in[15];
    const float* gv    = (const float*)d_in[16];
    const float* bev   = (const float*)d_in[17];
    const float* Wv2   = (const float*)d_in[18];
    const float* bv2   = (const float*)d_in[19];
    const float* Wa1   = (const float*)d_in[20];
    const float* ba1   = (const float*)d_in[21];
    const float* ga    = (const float*)d_in[22];
    const float* bea   = (const float*)d_in[23];
    const float* Wa2   = (const float*)d_in[24];
    const float* ba2   = (const float*)d_in[25];
    const int*   ev    = (const int*)d_in[26];

    // workspace layout (f16 halves)
    _Float16* ws   = (_Float16*)d_ws;
    _Float16* W1p  = ws + 0;        // 8*16*512 = 65536
    _Float16* W2p  = ws + 65536;    // 65536
    _Float16* W3p  = ws + 131072;   // 8*8*512  = 32768
    _Float16* Wv1p = ws + 163840;   // 4*4*512  = 8192
    _Float16* Wa1p = ws + 172032;   // 3 * 8192 = 24576
    _Float16* Wa2p = ws + 196608;   // 3 * 2048 = 6144

    auto pack = [&](const float* W, _Float16* dst, int KT, int NT, int N,
                    int realK, int strideW, int nb) {
        const int total = KT * NT * 512 * nb;
        pack_b_frags<<<(total + 255) / 256, 256, 0, stream>>>(
            W, dst, KT, NT, N, realK, strideW, nb);
    };
    pack(W1,  W1p,  8, 16, 256, 200, 200 * 256, 1);
    pack(W2,  W2p,  8, 16, 256, 256, 256 * 256, 1);
    pack(W3,  W3p,  8,  8, 128, 256, 256 * 128, 1);
    pack(Wv1, Wv1p, 4,  4,  64, 128, 128 *  64, 1);
    pack(Wa1, Wa1p, 4,  4,  64, 128, 128 *  64, NE);
    pack(Wa2, Wa2p, 2,  2,  32,  64,  64 *  32, NE);

    dueling_fused<<<NB / 64, 128, 0, stream>>>(
        state, td, b1, g1, be1, b2, g2, be2, b3, g3, be3,
        bv1, gv, bev, Wv2, bv2, ba1, ga, bea, ba2, ev,
        W1p, W2p, W3p, Wv1p, Wa1p, Wa2p, (float*)d_out);
}
